// Transformer_35476429865975
// MI455X (gfx1250) — compile-verified, matrix operand order checked by reading, not run on hardware
//
#include <hip/hip_runtime.h>
#include <math.h>
#include <stdint.h>

// ---------------- problem constants ----------------
constexpr int B_ = 2, S_ = 1024, L_ = 8, D_ = 768, F_ = 3072, V_ = 32000;
constexpr int H_ = 12;
constexpr int HD_ = D_ / H_;      // 64
constexpr int M_ = B_ * S_;       // 2048 token rows

// ---------------- vector types -----------------------
typedef __attribute__((ext_vector_type(8)))  __bf16 v8bf;
typedef __attribute__((ext_vector_type(16))) __bf16 v16bf;
typedef __attribute__((ext_vector_type(8)))  float  v8f;
typedef __attribute__((ext_vector_type(4)))  unsigned int v4u;
typedef __attribute__((ext_vector_type(8)))  int v8i;
typedef __attribute__((ext_vector_type(4)))  int v4i;

__device__ inline __bf16 tobf(float f) { return (__bf16)f; }

// Build a 16-element bf16 fragment from two contiguous 16B LDS chunks.
// CDNA5 16-bit A 16x32 / B 32x16 layout: element e holds
// K = (e&7) + 8*(lane>=16) + 16*(e>=8).
__device__ inline v16bf frag16(const __bf16* p0, const __bf16* p1) {
  v8bf a = *(const v8bf*)p0;
  v8bf b = *(const v8bf*)p1;
  v16bf f;
#pragma unroll
  for (int i = 0; i < 8; ++i) { f[i] = a[i]; f[i + 8] = b[i]; }
  return f;
}

__device__ inline v8f wmma_bf16(v16bf a, v16bf b, v8f c) {
  return __builtin_amdgcn_wmma_f32_16x16x32_bf16(false, a, false, b, (short)0, c,
                                                 false, false);
}

// ---------------- TDM: 2D bf16 tile -> LDS, padded rows ----------------
// Tile = tile_y rows x 32 bf16 (64B) per row, row stride = stride_elems.
// pad_enable, pad_interval=16 DWORDs (value 3), pad_amount=4 DWORDs (value 3)
// => LDS row stride 40 bf16 (80B), matching the fragment-load layout.
__device__ inline void tdm_load_tile32(unsigned lds_byte_off, const __bf16* gptr,
                                       unsigned tile_y, unsigned stride_elems) {
  unsigned long long ga = (unsigned long long)(uintptr_t)gptr;
  v4u g0;
  g0[0] = 1u;                                   // count=1 descriptor, user mode
  g0[1] = lds_byte_off;                         // lds_addr
  g0[2] = (unsigned)ga;                         // global_addr[31:0]
  g0[3] = (unsigned)((ga >> 32) & 0x1FFFFFFu) | (2u << 30);  // addr[56:32], type=2
  unsigned td0 = stride_elems;                  // tensor_dim0 >= tile_x (no OOB)
  unsigned td1 = 1u << 20;                      // tensor_dim1 large
  unsigned w0 = (1u << 16)                      // data_size = 2 bytes
              | (1u << 20)                      // pad_enable
              | (3u << 22)                      // pad_interval: 16 DWORDs
              | (3u << 25);                     // pad_amount:   4 DWORDs
  v8i g1;
  g1[0] = (int)w0;
  g1[1] = (int)((td0 & 0xFFFFu) << 16);         // tensor_dim0[15:0] (bits 63:48)
  g1[2] = (int)((td0 >> 16) | ((td1 & 0xFFFFu) << 16));
  g1[3] = (int)((td1 >> 16) | (32u << 16));     // tile_dim0 = 32 elements
  g1[4] = (int)(tile_y & 0xFFFFu);              // tile_dim1
  g1[5] = (int)stride_elems;                    // tensor_dim0_stride[31:0]
  g1[6] = 0;
  g1[7] = 0;
  v4i gz;
  gz[0] = 0; gz[1] = 0; gz[2] = 0; gz[3] = 0;
  v8i gz8;
#pragma unroll
  for (int i = 0; i < 8; ++i) gz8[i] = 0;
  // amdgpu-toolchain (clang-23) 6-arg form:
  // (uint32x4 g0, int32x8 g1, int32x4, int32x4, int32x8, i32 cpol)
  __builtin_amdgcn_tensor_load_to_lds(g0, g1, gz, gz, gz8, 0);
}

// ---------------- embedding + sinusoidal PE (fp32 residual stream) -------
__global__ __launch_bounds__(256) void embed_kernel(const int* __restrict__ idx,
                                                    const float* __restrict__ wte,
                                                    float* __restrict__ x) {
  int bs = blockIdx.x;
  int s = bs % S_;
  int token = idx[bs];
  float pos = (float)s;
  const float c = 9.210340371976184f / (float)D_;  // ln(10000)/D
  for (int d = threadIdx.x; d < D_; d += 256) {
    int i2 = (d >> 1) << 1;
    float freq = __expf(-(float)i2 * c);
    float pe = (d & 1) ? __cosf(pos * freq) : __sinf(pos * freq);
    x[(size_t)bs * D_ + d] = wte[(size_t)token * D_ + d] + pe;
  }
}

// ---------------- LayerNorm: fp32 in -> bf16 out ----------------
__global__ __launch_bounds__(256) void layernorm_kernel(const float* __restrict__ x,
                                                        const float* __restrict__ g,
                                                        const float* __restrict__ bta,
                                                        __bf16* __restrict__ out) {
  __shared__ float r1[256];
  __shared__ float r2[256];
  int row = blockIdx.x;
  int tid = threadIdx.x;
  const float* xr = x + (size_t)row * D_;
  float s = 0.f, s2 = 0.f;
  for (int d = tid; d < D_; d += 256) {
    float v = xr[d];
    s += v;
    s2 += v * v;
  }
  r1[tid] = s;
  r2[tid] = s2;
  __syncthreads();
  for (int o = 128; o > 0; o >>= 1) {
    if (tid < o) {
      r1[tid] += r1[tid + o];
      r2[tid] += r2[tid + o];
    }
    __syncthreads();
  }
  float mean = r1[0] * (1.0f / D_);
  float var = r2[0] * (1.0f / D_) - mean * mean;
  float rstd = rsqrtf(var + 1e-5f);
  __bf16* orow = out + (size_t)row * D_;
  for (int d = tid; d < D_; d += 256)
    orow[d] = tobf((xr[d] - mean) * rstd * g[d] + bta[d]);
}

// ---------------- weight transpose + convert: W[K,N] f32 -> WT[N,K] bf16 --
__global__ __launch_bounds__(256) void transpose_cvt_kernel(
    const float* __restrict__ W, __bf16* __restrict__ WT, int K, int N) {
  __shared__ float tile[32][33];
  const float* Wl = W + (size_t)blockIdx.z * K * N;
  __bf16* WTl = WT + (size_t)blockIdx.z * N * K;
  int tx = threadIdx.x & 31, ty = threadIdx.x >> 5;  // ty 0..7
  int k0 = blockIdx.y * 32, n0 = blockIdx.x * 32;
#pragma unroll
  for (int i = 0; i < 32; i += 8)
    tile[ty + i][tx] = Wl[(size_t)(k0 + ty + i) * N + n0 + tx];
  __syncthreads();
#pragma unroll
  for (int i = 0; i < 32; i += 8)
    WTl[(size_t)(n0 + ty + i) * K + k0 + tx] = tobf(tile[tx][ty + i]);
}

// ---------------- WMMA GEMM with TDM double-buffered staging ----------------
// A: [M,K] bf16, WT: [N,K] bf16 (pre-transposed). Tiles BM=64, BN=128, BK=32.
// Wave 0 drives the Tensor Data Mover; everyone else rides the barrier.
// ACT: 0 none, 1 exact GELU.  RES: add fp32 residual.  OUTBF: bf16 vs fp32 out.
template <int ACT, bool RES, bool OUTBF>
__global__ __launch_bounds__(256) void gemm_wmma_tdm(
    const __bf16* __restrict__ A, const __bf16* __restrict__ WT,
    const float* __restrict__ bias, const float* __restrict__ residual,
    void* __restrict__ Cout, int Ndim, int Kdim) {
  constexpr int BM = 64, BN = 128, BK = 32;
  constexpr int LDT = BK + 8;                 // 40 halfs (TDM pad -> 80B rows)
  __shared__ __bf16 lds_a[2][BM * LDT];
  __shared__ __bf16 lds_b[2][BN * LDT];

  int tid = threadIdx.x;
  int lane = tid & 31;
  int wv = tid >> 5;
  int wr = wv & 3, wc = wv >> 2;
  int hl = lane >> 4;
  int l16 = lane & 15;
  int m0 = blockIdx.y * BM;
  int n0 = blockIdx.x * BN;

  v8f acc[4];
#pragma unroll
  for (int t = 0; t < 4; ++t)
#pragma unroll
    for (int i = 0; i < 8; ++i) acc[t][i] = 0.f;

  const int nsteps = Kdim / BK;
  if (tid < 32) {  // wave 0: prologue DMA into buffer 0
    tdm_load_tile32((unsigned)(uintptr_t)&lds_a[0][0], A + (size_t)m0 * Kdim,
                    BM, (unsigned)Kdim);
    tdm_load_tile32((unsigned)(uintptr_t)&lds_b[0][0], WT + (size_t)n0 * Kdim,
                    BN, (unsigned)Kdim);
  }

  for (int step = 0; step < nsteps; ++step) {
    int cur = step & 1;
    if (tid < 32) __builtin_amdgcn_s_wait_tensorcnt(0);
    __syncthreads();  // staged tile visible to all waves
    if (step + 1 < nsteps && tid < 32) {  // overlap next DMA with compute
      int k1 = (step + 1) * BK;
      tdm_load_tile32((unsigned)(uintptr_t)&lds_a[1 - cur][0],
                      A + (size_t)m0 * Kdim + k1, BM, (unsigned)Kdim);
      tdm_load_tile32((unsigned)(uintptr_t)&lds_b[1 - cur][0],
                      WT + (size_t)n0 * Kdim + k1, BN, (unsigned)Kdim);
    }

    const __bf16* arow = &lds_a[cur][(wr * 16 + l16) * LDT];
    v16bf af = frag16(arow + 8 * hl, arow + 16 + 8 * hl);
#pragma unroll
    for (int t = 0; t < 4; ++t) {
      const __bf16* brow = &lds_b[cur][(wc * 64 + t * 16 + l16) * LDT];
      v16bf bf = frag16(brow + 8 * hl, brow + 16 + 8 * hl);
      acc[t] = wmma_bf16(af, bf, acc[t]);
    }
    __syncthreads();  // everyone done with buf[cur] before it is re-filled
  }

  // --- epilogue (fully branch-free per template instantiation) ---
#pragma unroll
  for (int t = 0; t < 4; ++t) {
    int n = n0 + wc * 64 + t * 16 + l16;
    float bv = bias[n];
#pragma unroll
    for (int r = 0; r < 8; ++r) {
      int m = m0 + wr * 16 + r + 8 * hl;
      float v = acc[t][r] + bv;
      if constexpr (ACT == 1) v = 0.5f * v * (1.f + erff(v * 0.70710678118f));
      if constexpr (RES) v += residual[(size_t)m * Ndim + n];
      if constexpr (OUTBF)
        ((__bf16*)Cout)[(size_t)m * Ndim + n] = tobf(v);
      else
        ((float*)Cout)[(size_t)m * Ndim + n] = v;
    }
  }
}

// ---------------- fused causal flash-attention (bf16 qkv in/out) ----------
// qkv: [B*S, 3*D] bf16, col = which*768 + h*64 + hd. attn: [B*S, D] bf16.
__global__ __launch_bounds__(128) void attention_kernel(
    const __bf16* __restrict__ qkv, __bf16* __restrict__ attn) {
  constexpr int LDK = HD_ + 8;  // 72 halfs, K stored [key][hd]
  constexpr int LDV = 32 + 8;   // 40 halfs, V stored transposed [hd][key]
  constexpr int LDP = 32 + 8;
  __shared__ __bf16 lds_k[32 * LDK];
  __shared__ __bf16 lds_v[HD_ * LDV];
  __shared__ __bf16 lds_p[4 * 16 * LDP];

  int tid = threadIdx.x;
  int lane = tid & 31;
  int wv = tid >> 5;
  int hl = lane >> 4;
  int l16 = lane & 15;
  int bh = blockIdx.y;
  int b = bh / H_, h = bh % H_;
  int q0blk = blockIdx.x * 64;
  int q0 = q0blk + wv * 16;

  // --- Q fragments straight from global bf16 ---
  v16bf aq[2];
  {
    const __bf16* qrow = qkv + (size_t)(b * S_ + q0 + l16) * (3 * D_) + h * HD_;
#pragma unroll
    for (int kh = 0; kh < 2; ++kh) {
#pragma unroll
      for (int c = 0; c < 2; ++c) {
        v8bf q = *(const v8bf*)(qrow + kh * 32 + c * 16 + hl * 8);
#pragma unroll
        for (int i = 0; i < 8; ++i) aq[kh][c * 8 + i] = q[i];
      }
    }
  }

  v8f o[4];
#pragma unroll
  for (int t = 0; t < 4; ++t)
#pragma unroll
    for (int i = 0; i < 8; ++i) o[t][i] = 0.f;
  float mprev[8], lsum[8];
#pragma unroll
  for (int r = 0; r < 8; ++r) { mprev[r] = -1e30f; lsum[r] = 0.f; }
  const float qscale = 0.125f;  // 1/sqrt(64)

  int jmax = q0blk + 63;
  for (int j0 = 0; j0 <= jmax; j0 += 32) {
    // --- cooperative K/V staging (pure bf16 copies) ---
#pragma unroll
    for (int i = 0; i < 2; ++i) {
      int c = tid + 128 * i;    // 0..255 chunks of 8 bf16
      int key = c >> 3;         // 0..31
      int hd = (c & 7) * 8;
      size_t base = (size_t)(b * S_ + j0 + key) * (3 * D_) + h * HD_ + hd;
      v8bf kv = *(const v8bf*)(qkv + base + D_);
      *(v8bf*)&lds_k[key * LDK + hd] = kv;
      v8bf vvv = *(const v8bf*)(qkv + base + 2 * D_);
#pragma unroll
      for (int j = 0; j < 8; ++j) lds_v[(hd + j) * LDV + key] = vvv[j];
    }
    __syncthreads();

    // --- S = Q K^T (two 16-key tiles, k-dim = HD = 2x32) ---
    v8f s[2];
#pragma unroll
    for (int t = 0; t < 2; ++t)
#pragma unroll
      for (int i = 0; i < 8; ++i) s[t][i] = 0.f;
#pragma unroll
    for (int kh = 0; kh < 2; ++kh) {
#pragma unroll
      for (int t = 0; t < 2; ++t) {
        const __bf16* krow = &lds_k[(t * 16 + l16) * LDK + kh * 32];
        v16bf bk = frag16(krow + 8 * hl, krow + 16 + 8 * hl);
        s[t] = wmma_bf16(aq[kh], bk, s[t]);
      }
    }

    // --- scale + causal mask in C-fragment layout ---
#pragma unroll
    for (int t = 0; t < 2; ++t) {
      int key = j0 + t * 16 + l16;
#pragma unroll
      for (int r = 0; r < 8; ++r) {
        int qr = q0 + r + 8 * hl;
        float sv = s[t][r] * qscale;
        s[t][r] = (key <= qr) ? sv : -1e9f;
      }
    }

    // --- online softmax ---
#pragma unroll
    for (int r = 0; r < 8; ++r) {
      float mx = fmaxf(s[0][r], s[1][r]);
#pragma unroll
      for (int msk = 8; msk >= 1; msk >>= 1) mx = fmaxf(mx, __shfl_xor(mx, msk, 32));
      float mnew = fmaxf(mprev[r], mx);
      float scale = __expf(mprev[r] - mnew);
      float e0 = __expf(s[0][r] - mnew);
      float e1 = __expf(s[1][r] - mnew);
      float rs = e0 + e1;
#pragma unroll
      for (int msk = 8; msk >= 1; msk >>= 1) rs += __shfl_xor(rs, msk, 32);
      lsum[r] = lsum[r] * scale + rs;
      mprev[r] = mnew;
#pragma unroll
      for (int t = 0; t < 4; ++t) o[t][r] *= scale;
      s[0][r] = e0;
      s[1][r] = e1;
    }

    // --- P (C-layout) -> A-fragment via per-wave LDS tile ---
    __bf16* pw = &lds_p[wv * 16 * LDP];
#pragma unroll
    for (int r = 0; r < 8; ++r) {
      pw[(r + 8 * hl) * LDP + l16] = tobf(s[0][r]);
      pw[(r + 8 * hl) * LDP + 16 + l16] = tobf(s[1][r]);
    }
    __builtin_amdgcn_wave_barrier();
    asm volatile("s_wait_dscnt 0x0" ::: "memory");
    const __bf16* prow = &pw[l16 * LDP];
    v16bf ap = frag16(prow + 8 * hl, prow + 16 + 8 * hl);

    // --- O += P V ---
#pragma unroll
    for (int t = 0; t < 4; ++t) {
      const __bf16* vrow = &lds_v[(t * 16 + l16) * LDV];
      v16bf bv = frag16(vrow + 8 * hl, vrow + 16 + 8 * hl);
      o[t] = wmma_bf16(ap, bv, o[t]);
    }
    __syncthreads();
  }

  // --- finalize: O / l -> bf16 ---
#pragma unroll
  for (int r = 0; r < 8; ++r) {
    float inv = 1.0f / lsum[r];
    int m = q0 + r + 8 * hl;
    __bf16* orow = attn + (size_t)(b * S_ + m) * D_ + h * HD_;
#pragma unroll
    for (int t = 0; t < 4; ++t) orow[t * 16 + l16] = tobf(o[t][r] * inv);
  }
}

// ---------------- loss ----------------
__global__ void zero_loss_kernel(float* loss) {
  if (threadIdx.x == 0 && blockIdx.x == 0) *loss = 0.f;
}

__global__ __launch_bounds__(256) void loss_kernel(const float* __restrict__ logits,
                                                   const int* __restrict__ targets,
                                                   float* __restrict__ loss) {
  __shared__ float red[256];
  int row = blockIdx.x;
  int tid = threadIdx.x;
  const float* lr = logits + (size_t)row * V_;
  float mx = -1e30f;
  for (int i = tid; i < V_; i += 256) mx = fmaxf(mx, lr[i]);
  red[tid] = mx;
  __syncthreads();
  for (int o = 128; o > 0; o >>= 1) {
    if (tid < o) red[tid] = fmaxf(red[tid], red[tid + o]);
    __syncthreads();
  }
  mx = red[0];
  __syncthreads();
  float s = 0.f;
  for (int i = tid; i < V_; i += 256) s += __expf(lr[i] - mx);
  red[tid] = s;
  __syncthreads();
  for (int o = 128; o > 0; o >>= 1) {
    if (tid < o) red[tid] += red[tid + o];
    __syncthreads();
  }
  if (tid == 0) {
    float lse = mx + logf(red[0]);
    float lp = lr[targets[row]] - lse;
    atomicAdd(loss, -lp / (float)(B_ * S_));
  }
}

// ---------------- host orchestration ----------------
extern "C" void kernel_launch(void* const* d_in, const int* in_sizes, int n_in,
                              void* d_out, int out_size, void* d_ws, size_t ws_size,
                              hipStream_t stream) {
  (void)in_sizes; (void)n_in; (void)out_size; (void)ws_size;
  const int* idx = (const int*)d_in[0];
  const int* targets = (const int*)d_in[1];
  const float* wte = (const float*)d_in[2];
  const float* qkv_w = (const float*)d_in[3];
  const float* qkv_b = (const float*)d_in[4];
  const float* out_w = (const float*)d_in[5];
  const float* out_b = (const float*)d_in[6];
  const float* ffn_w1 = (const float*)d_in[7];
  const float* ffn_b1 = (const float*)d_in[8];
  const float* ffn_w2 = (const float*)d_in[9];
  const float* ffn_b2 = (const float*)d_in[10];
  const float* ln1_g = (const float*)d_in[11];
  const float* ln1_b = (const float*)d_in[12];
  const float* ln2_g = (const float*)d_in[13];
  const float* ln2_b = (const float*)d_in[14];
  const float* lnf_g = (const float*)d_in[15];
  const float* lnf_b = (const float*)d_in[16];
  const float* head_w = (const float*)d_in[17];
  const float* head_b = (const float*)d_in[18];

  float* logits = (float*)d_out;
  float* loss = logits + (size_t)M_ * V_;

  char* p = (char*)d_ws;
  auto carve = [&](size_t bytes) {
    char* r = p;
    p += (bytes + 255) & ~(size_t)255;
    return r;
  };
  float* x = (float*)carve((size_t)M_ * D_ * 4);
  __bf16* hbuf = (__bf16*)carve((size_t)M_ * D_ * 2);
  __bf16* qkvb = (__bf16*)carve((size_t)M_ * 3 * D_ * 2);
  __bf16* attnb = (__bf16*)carve((size_t)M_ * D_ * 2);
  __bf16* ffnh = (__bf16*)carve((size_t)M_ * F_ * 2);
  __bf16* qkvT = (__bf16*)carve((size_t)L_ * 3 * D_ * D_ * 2);
  __bf16* outT = (__bf16*)carve((size_t)L_ * D_ * D_ * 2);
  __bf16* w1T = (__bf16*)carve((size_t)L_ * F_ * D_ * 2);
  __bf16* w2T = (__bf16*)carve((size_t)L_ * D_ * F_ * 2);
  __bf16* headT = (__bf16*)carve((size_t)V_ * D_ * 2);

  // ---- one-pass weight transpose+convert to bf16 [N][K] ----
  transpose_cvt_kernel<<<dim3(3 * D_ / 32, D_ / 32, L_), 256, 0, stream>>>(
      qkv_w, qkvT, D_, 3 * D_);
  transpose_cvt_kernel<<<dim3(D_ / 32, D_ / 32, L_), 256, 0, stream>>>(
      out_w, outT, D_, D_);
  transpose_cvt_kernel<<<dim3(F_ / 32, D_ / 32, L_), 256, 0, stream>>>(
      ffn_w1, w1T, D_, F_);
  transpose_cvt_kernel<<<dim3(D_ / 32, F_ / 32, L_), 256, 0, stream>>>(
      ffn_w2, w2T, F_, D_);
  transpose_cvt_kernel<<<dim3(V_ / 32, D_ / 32, 1), 256, 0, stream>>>(
      head_w, headT, D_, V_);

  embed_kernel<<<M_, 256, 0, stream>>>(idx, wte, x);

  for (int l = 0; l < L_; ++l) {
    layernorm_kernel<<<M_, 256, 0, stream>>>(x, ln1_g + l * D_, ln1_b + l * D_, hbuf);
    gemm_wmma_tdm<0, false, true><<<dim3(3 * D_ / 128, M_ / 64), 256, 0, stream>>>(
        hbuf, qkvT + (size_t)l * 3 * D_ * D_, qkv_b + (size_t)l * 3 * D_, nullptr,
        qkvb, 3 * D_, D_);
    attention_kernel<<<dim3(S_ / 64, B_ * H_), 128, 0, stream>>>(qkvb, attnb);
    gemm_wmma_tdm<0, true, false><<<dim3(D_ / 128, M_ / 64), 256, 0, stream>>>(
        attnb, outT + (size_t)l * D_ * D_, out_b + (size_t)l * D_, x, x, D_, D_);
    layernorm_kernel<<<M_, 256, 0, stream>>>(x, ln2_g + l * D_, ln2_b + l * D_, hbuf);
    gemm_wmma_tdm<1, false, true><<<dim3(F_ / 128, M_ / 64), 256, 0, stream>>>(
        hbuf, w1T + (size_t)l * F_ * D_, ffn_b1 + (size_t)l * F_, nullptr, ffnh,
        F_, D_);
    gemm_wmma_tdm<0, true, false><<<dim3(D_ / 128, M_ / 64), 256, 0, stream>>>(
        ffnh, w2T + (size_t)l * D_ * F_, ffn_b2 + (size_t)l * D_, x, x, D_, F_);
  }

  layernorm_kernel<<<M_, 256, 0, stream>>>(x, lnf_g, lnf_b, hbuf);
  gemm_wmma_tdm<0, false, false><<<dim3(V_ / 128, M_ / 64), 256, 0, stream>>>(
      hbuf, headT, head_b, nullptr, logits, V_, D_);
  zero_loss_kernel<<<1, 1, 0, stream>>>(loss);
  loss_kernel<<<M_, 256, 0, stream>>>(logits, targets, loss);
}
